// HopfRNNLayerTheta_80221399154894
// MI455X (gfx1250) — matrix operand order repaired and checked by reading, not compiled, not run on hardware
//
#include <hip/hip_runtime.h>

typedef __attribute__((ext_vector_type(16))) _Float16 v16h;
typedef __attribute__((ext_vector_type(8)))  float    v8f;
typedef __attribute__((ext_vector_type(4)))  float    fvec4;
typedef __attribute__((ext_vector_type(2)))  float    fvec2;

// ---------------------------------------------------------------------------
// K1: s[b,tt] = sum_v x[b,tt,v]   (dominant: streams 268 MB of x, read-once)
// One wave32 per 1024-float row; lanes do strided 16B loads (coalesced 512B
// per step) with NON-TEMPORAL hint so the one-shot stream of x does not
// evict the small reused intermediates from the 192 MB L2.
// ---------------------------------------------------------------------------
__global__ void k_rowsum(const float* __restrict__ x, float* __restrict__ s) {
    const int wave = threadIdx.x >> 5;
    const int lane = threadIdx.x & 31;
    const int row  = blockIdx.x * 8 + wave;              // 0..65535 (b*1024+tt)
    const fvec4* p = (const fvec4*)(x + (size_t)row * 1024);
    float acc = 0.f;
#pragma unroll
    for (int i = 0; i < 8; ++i) {
        fvec4 v = __builtin_nontemporal_load(&p[i * 32 + lane]);
        acc += v.x + v.y + v.z + v.w;
    }
#pragma unroll
    for (int off = 16; off > 0; off >>= 1)
        acc += __shfl_xor(acc, off, 32);                 // wave32 on gfx1250
    if (lane == 0) s[row] = acc;
}

// ---------------------------------------------------------------------------
// K2: DFT twiddles, written directly in WMMA B-fragment order (f16).
// B (32x16, KxN) wave32 layout: N = lane%16, K = (j&7) + 8*(2*(j>>3) + lane/16).
// Fragment index: [ntile][kstep][lane][j] ; value at tt=kstep*32+K,
// kt=ntile*16+(lane&15) ; phase index p=(tt*kt) mod 1024 kept exact in integer.
// ---------------------------------------------------------------------------
__global__ void k_twiddle(_Float16* __restrict__ Fc, _Float16* __restrict__ Fs) {
    const int gid   = blockIdx.x * 256 + threadIdx.x;    // 0..524287
    const int j     = gid & 15;
    const int lane  = (gid >> 4) & 31;
    const int kstep = (gid >> 9) & 31;
    const int ntile = gid >> 14;                         // 0..31
    const int tt = kstep * 32 + (j & 7) + 8 * ((j >> 3) * 2 + (lane >> 4));
    const int kt = ntile * 16 + (lane & 15);
    const int p  = (tt * kt) & 1023;
    const float th = (float)p * 6.135923151543e-3f;      // 2*pi/1024
    float sn, cs;
    __sincosf(th, &sn, &cs);
    Fc[gid] = (_Float16)cs;
    Fs[gid] = (_Float16)sn;
}

// ---------------------------------------------------------------------------
// K3: T1 = (0.5/1024) * Bm @ S  (complex 64x1024), written directly in WMMA
// A-fragment order (f16). A (16x32 f16) layout per ISA 7.12.2:
//  lane<16: K = j (j<8), j+8 (j>=8) ; lane>=16: K = j+8 (j<8), j+16 (j>=8)
// Fragment index: [mtile][kstep][lane][j] ; b = mtile*16 + (lane&15),
// tt = kstep*32 + K.
// ---------------------------------------------------------------------------
__global__ void k_t1(const float* __restrict__ s,
                     const float* __restrict__ Br, const float* __restrict__ Bi,
                     _Float16* __restrict__ T1r, _Float16* __restrict__ T1i) {
    const int gid   = blockIdx.x * 256 + threadIdx.x;    // 0..65535
    const int j     = gid & 15;
    const int lane  = (gid >> 4) & 31;
    const int kstep = (gid >> 9) & 31;
    const int mtile = gid >> 14;                         // 0..3
    const int hlf   = lane >> 4;
    const int b     = mtile * 16 + (lane & 15);
    const int K     = (j & 7) + 8 * ((j >> 3) * 2 + hlf);
    const int tt    = kstep * 32 + K;
    float ar = 0.f, ai = 0.f;
    for (int bp = 0; bp < 64; ++bp) {
        const float sv = s[bp * 1024 + tt];
        ar += Br[b * 64 + bp] * sv;
        ai += Bi[b * 64 + bp] * sv;
    }
    const float scale = 0.5f / 1024.0f;   // 0.5 * (1/1024^2 fftnorm^2) * 1024 out-scale
    T1r[gid] = (_Float16)(ar * scale);
    T1i[gid] = (_Float16)(ai * scale);
}

// ---------------------------------------------------------------------------
// K4: Z = T1 @ (Fc - i*Fs)  via v_wmma_f32_16x16x32_f16.
//   Zr = T1r@Fc + T1i@Fs ;  Zi = T1i@Fc - T1r@Fs
// f16 WMMA has no A-negate (NEG = {CNeg,0,0}), so accumulate T1r@Fs
// separately and subtract at the end. One wave per 16x16 output tile,
// K=1024 -> 32 k-steps x 4 wmma. 128 tiles total.
// ---------------------------------------------------------------------------
__global__ void k_wmma_dft(const _Float16* __restrict__ T1r, const _Float16* __restrict__ T1i,
                           const _Float16* __restrict__ Fc,  const _Float16* __restrict__ Fs,
                           float* __restrict__ Zr, float* __restrict__ Zi) {
    const int wave  = threadIdx.x >> 5;
    const int lane  = threadIdx.x & 31;
    const int tile  = blockIdx.x * 4 + wave;             // 0..127
    const int mtile = tile >> 5;                         // 0..3
    const int ntile = tile & 31;                         // 0..31
    v8f accR = {}, accIc = {}, accIs = {};
    for (int k = 0; k < 32; ++k) {
        const v16h ar = *(const v16h*)(T1r + (((mtile * 32 + k) * 32 + lane) << 4));
        const v16h ai = *(const v16h*)(T1i + (((mtile * 32 + k) * 32 + lane) << 4));
        const v16h bc = *(const v16h*)(Fc  + (((ntile * 32 + k) * 32 + lane) << 4));
        const v16h bs = *(const v16h*)(Fs  + (((ntile * 32 + k) * 32 + lane) << 4));
        accR  = __builtin_amdgcn_wmma_f32_16x16x32_f16(false, ar, false, bc, (short)0, accR,  false, false);
        accR  = __builtin_amdgcn_wmma_f32_16x16x32_f16(false, ai, false, bs, (short)0, accR,  false, false);
        accIc = __builtin_amdgcn_wmma_f32_16x16x32_f16(false, ai, false, bc, (short)0, accIc, false, false);
        accIs = __builtin_amdgcn_wmma_f32_16x16x32_f16(false, ar, false, bs, (short)0, accIs, false, false);
    }
    // C/D layout: VGPR r holds M = r + 8*(lane/16), N = lane%16
    const int n  = ntile * 16 + (lane & 15);
    const int mb = mtile * 16 + ((lane >> 4) << 3);
#pragma unroll
    for (int r = 0; r < 8; ++r) {
        Zr[(mb + r) * 512 + n] = accR[r];
        Zi[(mb + r) * 512 + n] = accIc[r] - accIs[r];
    }
}

// ---------------------------------------------------------------------------
// K5: out[b,u,t,{re,im}] = G[b,t] + (t+u>=511 ? G[b,t+u-511] : 0)
// (G = Z; the z-state head is always zero so this is the whole scan.)
// Block: one (b, 16-u chunk); G row cached in LDS; coalesced 8B stores with
// NON-TEMPORAL hint (134 MB written once, never re-read on device).
// ---------------------------------------------------------------------------
__global__ void k_out(const float* __restrict__ Zr, const float* __restrict__ Zi,
                      fvec2* __restrict__ out) {
    __shared__ float gr[512];
    __shared__ float gi[512];
    const int b  = blockIdx.x >> 5;                      // 0..63
    const int uc = blockIdx.x & 31;                      // 0..31
    for (int i = threadIdx.x; i < 512; i += 256) {
        gr[i] = Zr[b * 512 + i];
        gi[i] = Zi[b * 512 + i];
    }
    __syncthreads();
    for (int uu = 0; uu < 16; ++uu) {
        const int u = uc * 16 + uu;
        fvec2* row = out + ((size_t)(b * 512 + u)) * 512;
        for (int t = threadIdx.x; t < 512; t += 256) {
            const int d = t + u - 511;
            float re = gr[t], im = gi[t];
            if (d >= 0) { re += gr[d]; im += gi[d]; }
            fvec2 val;
            val.x = re;
            val.y = im;
            __builtin_nontemporal_store(val, &row[t]);
        }
    }
}

// ---------------------------------------------------------------------------
extern "C" void kernel_launch(void* const* d_in, const int* in_sizes, int n_in,
                              void* d_out, int out_size, void* d_ws, size_t ws_size,
                              hipStream_t stream) {
    const float* x  = (const float*)d_in[0];
    // d_in[1] = A_r, d_in[2] = A_i: provably unused — the scan reads only the
    // head of a 512-deep shift register over 512 steps, which is always zero.
    const float* Br = (const float*)d_in[3];
    const float* Bi = (const float*)d_in[4];

    char* ws = (char*)d_ws;
    float*    s   = (float*)   (ws + 0);         // 64*1024 f32      = 256 KB
    _Float16* T1r = (_Float16*)(ws + 262144);    // 64*1024 f16      = 128 KB
    _Float16* T1i = (_Float16*)(ws + 393216);    // 64*1024 f16      = 128 KB
    _Float16* Fc  = (_Float16*)(ws + 524288);    // 1024*512 f16     =   1 MB
    _Float16* Fs  = (_Float16*)(ws + 1572864);   // 1024*512 f16     =   1 MB
    float*    Zr  = (float*)   (ws + 2621440);   // 64*512 f32       = 128 KB
    float*    Zi  = (float*)   (ws + 2752512);   // 64*512 f32       = 128 KB

    hipLaunchKernelGGL(k_rowsum,   dim3(8192), dim3(256), 0, stream, x, s);
    hipLaunchKernelGGL(k_twiddle,  dim3(2048), dim3(256), 0, stream, Fc, Fs);
    hipLaunchKernelGGL(k_t1,       dim3(256),  dim3(256), 0, stream, s, Br, Bi, T1r, T1i);
    hipLaunchKernelGGL(k_wmma_dft, dim3(32),   dim3(128), 0, stream, T1r, T1i, Fc, Fs, Zr, Zi);
    hipLaunchKernelGGL(k_out,      dim3(2048), dim3(256), 0, stream, Zr, Zi, (fvec2*)d_out);
}
